// AxialAttention3D_80865644249162
// MI455X (gfx1250) — matrix operand order repaired and checked by reading, not compiled, not run on hardware
//
#include <hip/hip_runtime.h>
#include <hip/hip_bf16.h>

typedef __attribute__((ext_vector_type(16))) _Float16 v16h;
typedef __attribute__((ext_vector_type(8)))  _Float16 v8h;
typedef __attribute__((ext_vector_type(8)))  float    v8f;

#define CCH    256
#define SEQ    48
#define NHEAD  8
#define HD     32
#define STRC   110592L          // channel stride = 48*48*48
#define LN_EPS 1e-5f

// ---- LDS layout (bytes), total 229376 (<320KB/WGP) ----
#define OFF_XSEQ 0              // [48][256] f32  = 49152  (kept for residual)
#define OFF_XN   49152          // [48][256] f16  = 24576  (LN output; later aliased as O)
#define OFF_QK   73728          // [8][2][48][32] f16 = 49152 (Q,K; later aliased as P [8][48][64])
#define OFF_VT   122880         // [8][32][64] f16 = 32768 (V transposed, K-pad 48->64 zeroed)
#define OFF_SC   155648         // [8][48][48] f32 = 73728 (scores; later aliased as result [48][256] f32)
#define LDS_BYTES 229376

// ---------------------------------------------------------------------------
// WMMA fragment loaders (contiguous k order used by this code for both A and B, not the ISA table order)
// A 16x32 f16: lane l -> row = l&15, khalf = l>>4.
//   halves 0..15 = K (khalf*16 .. +15)        (contiguous 32B)
// B 32x16 f16: lane l -> col n = l&15, khalf = l>>4.
//   halves 0..15 = K (khalf*16 .. +15)        (contiguous 32B)
// B column n is stored as row n of a row-major matrix (W row, K row, Vt row).
// ---------------------------------------------------------------------------
__device__ __forceinline__ v16h load_a_frag(const _Float16* __restrict__ base,
                                            int rowStride, int row0, int k0, int lane) {
  const int r  = row0 + (lane & 15);
  const int kh = lane >> 4;
  const _Float16* p = base + r * rowStride + k0 + kh * 16;
  v8h lo = *(const v8h*)p;
  v8h hi = *(const v8h*)(p + 8);
  v16h f;
#pragma unroll
  for (int i = 0; i < 8; ++i) { f[i] = lo[i]; f[8 + i] = hi[i]; }
  return f;
}

__device__ __forceinline__ v16h load_b_frag(const _Float16* __restrict__ base,
                                            int rowStride, int n0, int k0, int lane) {
  const int n  = n0 + (lane & 15);
  const int kh = lane >> 4;
  const _Float16* p = base + n * rowStride + k0 + kh * 16;
  v8h lo = *(const v8h*)p;
  v8h hi = *(const v8h*)(p + 8);
  v16h f;
#pragma unroll
  for (int i = 0; i < 8; ++i) { f[i] = lo[i]; f[8 + i] = hi[i]; }
  return f;
}

__device__ __forceinline__ v8f wmma_f16(v16h a, v16h b, v8f c) {
  return __builtin_amdgcn_wmma_f32_16x16x32_f16(false, a, false, b, (short)0, c, false, false);
}

// generic LDS pointer -> hardware LDS byte offset (ISA §10.2: low 32 bits of the address)
__device__ __forceinline__ unsigned lds_off(const void* p) {
  return (unsigned)(uintptr_t)p;
}

// branchless Q/K/V routing of one 16x16 accumulator tile into LDS
__device__ __forceinline__ void qkv_store(_Float16* lds16, const float* __restrict__ bqkv,
                                          v8f acc, int m0, int n0, int lane) {
  const int   n     = n0 + (lane & 15);
  const float bi    = bqkv[n];
  const int   part  = n >> 8;               // 0=Q 1=K 2=V
  const int   nn    = n & 255;
  const int   h     = nn >> 5, d = nn & 31;
  const int   isV   = part >> 1;
  const float scale = (part == 0) ? 0.17677669529663687f : 1.0f;  // 1/sqrt(32) into Q
  const int   aQK   = OFF_QK / 2 + ((h * 2 + part) * SEQ) * HD + d;  // + m*HD
  const int   aV    = OFF_VT / 2 + (h * HD + d) * 64;                // + m
#pragma unroll
  for (int rr = 0; rr < 8; ++rr) {
    const int m    = m0 + (lane >> 4) * 8 + rr;
    const int addr = isV ? (aV + m) : (aQK + m * HD);
    lds16[addr] = (_Float16)((acc[rr] + bi) * scale);
  }
}

// fp32 -> fp16 weight repack (weights tiny: 1.5MB total, L2-resident afterwards)
extern "C" __global__ void repack_f16(const float* __restrict__ src,
                                      _Float16* __restrict__ dst, int n) {
  int i = blockIdx.x * blockDim.x + threadIdx.x;
  if (i < n) dst[i] = (_Float16)src[i];
}

// ---------------------------------------------------------------------------
// One axial pass: one block per sequence (4608 blocks), 256 threads = 8 waves.
// ---------------------------------------------------------------------------
extern "C" __global__ __launch_bounds__(256)
void axial_pass(const float* __restrict__ src, float* __restrict__ dst,
                const float* __restrict__ gamma,
                const float* __restrict__ lnw, const float* __restrict__ lnb,
                const _Float16* __restrict__ wqkv, const float* __restrict__ bqkv,
                const _Float16* __restrict__ wo,   const float* __restrict__ bo,
                long strideT, long strideA, long strideB) {
  extern __shared__ char smem[];
  float*    xseq  = (float*)(smem + OFF_XSEQ);      // [48][256]
  _Float16* xn    = (_Float16*)(smem + OFF_XN);     // [48][256] ; aliased as O later
  _Float16* qk    = (_Float16*)(smem + OFF_QK);     // [8][2][48][32]; aliased as P[8][48][64]
  _Float16* vt    = (_Float16*)(smem + OFF_VT);     // [8][32][64]
  float*    sc    = (float*)(smem + OFF_SC);        // [8][48][48]
  float*    res   = (float*)(smem + OFF_SC);        // [48][256] result (aliases sc, dead by then)
  _Float16* lds16 = (_Float16*)smem;                // flat half view for branchless scatter

  const int tid  = threadIdx.x;
  const int lane = tid & 31;
  const int wave = tid >> 5;

  const int s = blockIdx.x;
  const int b = s / (48 * 48);
  const int r = s % (48 * 48);
  const long base = (long)b * CCH * STRC + (long)(r / 48) * strideA + (long)(r % 48) * strideB;
  const float* srcBase = src + base;
  float*       dstBase = dst + base;

  // ---- phase 0: async-gather sequence into LDS (token-minor: best coalescing
  //      for every pass since strideT <= strideC always); zero Vt K-pad ----
  for (int idx = tid; idx < SEQ * CCH; idx += 256) {
    const int m = idx % SEQ, c = idx / SEQ;
    const unsigned goff = (unsigned)(((long)m * strideT + (long)c * STRC) * 4);
    const unsigned loff = lds_off(&xseq[m * CCH + c]);
    asm volatile("global_load_async_to_lds_b32 %0, %1, %2"
                 :: "v"(loff), "v"(goff), "s"(srcBase) : "memory");
  }
  for (int idx = tid; idx < (NHEAD * HD * 64) / 2; idx += 256) ((int*)vt)[idx] = 0;
  asm volatile("s_wait_asynccnt 0x0" ::: "memory");
  __syncthreads();

  // ---- phase 1: LayerNorm -> xn (f16). One wave per token. ----
  const float g = gamma[0];
  for (int m = wave; m < SEQ; m += 8) {
    float s0 = 0.f;
#pragma unroll
    for (int i = 0; i < 8; ++i) s0 += xseq[m * CCH + lane + 32 * i];
#pragma unroll
    for (int off = 16; off > 0; off >>= 1) s0 += __shfl_xor(s0, off, 32);
    const float mu = s0 * (1.0f / 256.0f);
    float v0 = 0.f;
#pragma unroll
    for (int i = 0; i < 8; ++i) {
      float d = xseq[m * CCH + lane + 32 * i] - mu;
      v0 += d * d;
    }
#pragma unroll
    for (int off = 16; off > 0; off >>= 1) v0 += __shfl_xor(v0, off, 32);
    const float inv = rsqrtf(v0 * (1.0f / 256.0f) + LN_EPS);
#pragma unroll
    for (int i = 0; i < 8; ++i) {
      int c = lane + 32 * i;
      xn[m * CCH + c] = (_Float16)((xseq[m * CCH + c] - mu) * inv * lnw[c] + lnb[c]);
    }
  }
  __syncthreads();

  // ---- phase 2: QKV = xn @ Wqkv^T + b.
  //      72 dual-N tile pairs (9/wave, exact): one shared A fragment feeds two
  //      independent WMMA chains; double-buffered fragment loads so waits
  //      before each WMMA only cover loads issued one k-step earlier. ----
  for (int t = wave; t < 72; t += 8) {
    const int mt = t % 3, ntp = t / 3;
    const int m0 = mt * 16, n0 = ntp * 32;
    __builtin_prefetch(&wqkv[(n0 + (lane & 15)) * CCH], 0, 3);
    v16h a0 = load_a_frag(xn, CCH, m0, 0, lane);
    v16h p0 = load_b_frag(wqkv, CCH, n0, 0, lane);
    v16h q0 = load_b_frag(wqkv, CCH, n0 + 16, 0, lane);
    v8f acc0 = {}, acc1 = {};
#pragma unroll
    for (int k = 0; k < 7; ++k) {
      v16h a1 = load_a_frag(xn, CCH, m0, (k + 1) * 32, lane);
      v16h p1 = load_b_frag(wqkv, CCH, n0, (k + 1) * 32, lane);
      v16h q1 = load_b_frag(wqkv, CCH, n0 + 16, (k + 1) * 32, lane);
      acc0 = wmma_f16(a0, p0, acc0);
      acc1 = wmma_f16(a0, q0, acc1);
      a0 = a1; p0 = p1; q0 = q1;
    }
    acc0 = wmma_f16(a0, p0, acc0);
    acc1 = wmma_f16(a0, q0, acc1);
    qkv_store(lds16, bqkv, acc0, m0, n0, lane);
    qkv_store(lds16, bqkv, acc1, m0, n0 + 16, lane);
  }
  __syncthreads();

  // ---- phase 3: per-head attention. head == wave (8/8). ----
  {
    const int h = wave;
    const _Float16* qh  = qk + (h * 2 + 0) * SEQ * HD;    // [48][32]
    const _Float16* kh  = qk + (h * 2 + 1) * SEQ * HD;    // [48][32]
    float* sch = sc + h * SEQ * SEQ;                      // [48][48]

    // scores = (Q/sqrt(hd)) @ K^T : single K-step (hd=32)
#pragma unroll
    for (int mt = 0; mt < 3; ++mt)
#pragma unroll
      for (int nt = 0; nt < 3; ++nt) {
        v8f acc = {};
        v16h a  = load_a_frag(qh, HD, mt * 16, 0, lane);
        v16h bf = load_b_frag(kh, HD, nt * 16, 0, lane);
        acc = wmma_f16(a, bf, acc);
        const int n = nt * 16 + (lane & 15);
#pragma unroll
        for (int rr = 0; rr < 8; ++rr)
          sch[(mt * 16 + (lane >> 4) * 8 + rr) * SEQ + n] = acc[rr];
      }

    // softmax rows -> P f16 [48][64] (cols 48..63 zero), aliased over this head's Q/K
    _Float16* ph = qk + h * 2 * SEQ * HD;
    for (int row = lane; row < SEQ; row += 32) {
      float mx = -1e30f;
      for (int c = 0; c < SEQ; ++c) mx = fmaxf(mx, sch[row * SEQ + c]);
      float sum = 0.f;
      for (int c = 0; c < SEQ; ++c) sum += __expf(sch[row * SEQ + c] - mx);
      const float rinv = 1.0f / sum;
      for (int c = 0; c < SEQ; ++c)
        ph[row * 64 + c] = (_Float16)(__expf(sch[row * SEQ + c] - mx) * rinv);
      for (int c = SEQ; c < 64; ++c) ph[row * 64 + c] = (_Float16)0.0f;
    }

    // O_head = P @ V : A = P [48][64], B col n = Vt row n ; 3 M-tiles x dual-N x 2 k-steps
    const _Float16* vth  = vt + h * HD * 64;              // [32][64]
    _Float16*       obuf = xn;                            // alias (xn dead after phase 2)
#pragma unroll
    for (int mt = 0; mt < 3; ++mt) {
      v8f acc0 = {}, acc1 = {};
#pragma unroll
      for (int k = 0; k < 2; ++k) {
        v16h a  = load_a_frag(ph, 64, mt * 16, k * 32, lane);
        v16h b0 = load_b_frag(vth, 64, 0, k * 32, lane);
        v16h b1 = load_b_frag(vth, 64, 16, k * 32, lane);
        acc0 = wmma_f16(a, b0, acc0);
        acc1 = wmma_f16(a, b1, acc1);
      }
      const int n = lane & 15;
#pragma unroll
      for (int rr = 0; rr < 8; ++rr) {
        const int m = mt * 16 + (lane >> 4) * 8 + rr;
        obuf[m * CCH + h * HD + n]      = (_Float16)acc0[rr];
        obuf[m * CCH + h * HD + 16 + n] = (_Float16)acc1[rr];
      }
    }
  }
  __syncthreads();

  // ---- phase 4: res = xseq + gamma * (O @ Wo^T + bo), staged in LDS.
  //      24 dual-N tile pairs (3/wave, exact), double-buffered. ----
  const _Float16* obuf = xn;
  for (int t = wave; t < 24; t += 8) {
    const int mt = t % 3, ntp = t / 3;
    const int m0 = mt * 16, n0 = ntp * 32;
    __builtin_prefetch(&wo[(n0 + (lane & 15)) * CCH], 0, 3);
    v16h a0 = load_a_frag(obuf, CCH, m0, 0, lane);
    v16h p0 = load_b_frag(wo, CCH, n0, 0, lane);
    v16h q0 = load_b_frag(wo, CCH, n0 + 16, 0, lane);
    v8f acc0 = {}, acc1 = {};
#pragma unroll
    for (int k = 0; k < 7; ++k) {
      v16h a1 = load_a_frag(obuf, CCH, m0, (k + 1) * 32, lane);
      v16h p1 = load_b_frag(wo, CCH, n0, (k + 1) * 32, lane);
      v16h q1 = load_b_frag(wo, CCH, n0 + 16, (k + 1) * 32, lane);
      acc0 = wmma_f16(a0, p0, acc0);
      acc1 = wmma_f16(a0, q0, acc1);
      a0 = a1; p0 = p1; q0 = q1;
    }
    acc0 = wmma_f16(a0, p0, acc0);
    acc1 = wmma_f16(a0, q0, acc1);
#pragma unroll
    for (int half = 0; half < 2; ++half) {
      const v8f&  acc = half ? acc1 : acc0;
      const int   n   = n0 + half * 16 + (lane & 15);
      const float bi  = bo[n];
#pragma unroll
      for (int rr = 0; rr < 8; ++rr) {
        const int m = m0 + (lane >> 4) * 8 + rr;
        res[m * CCH + n] = xseq[m * CCH + n] + g * (acc[rr] + bi);
      }
    }
  }
  __syncthreads();

  // ---- phase 5: async-scatter result LDS -> global (token-minor mapping);
  //      S_ENDPGM's implicit wait-idle drains ASYNCcnt before block retires ----
  for (int idx = tid; idx < SEQ * CCH; idx += 256) {
    const int m = idx % SEQ, c = idx / SEQ;
    const unsigned goff = (unsigned)(((long)m * strideT + (long)c * STRC) * 4);
    const unsigned loff = lds_off(&res[m * CCH + c]);
    asm volatile("global_store_async_from_lds_b32 %0, %1, %2"
                 :: "v"(goff), "v"(loff), "s"(dstBase) : "memory");
  }
}

// ---------------------------------------------------------------------------
extern "C" void kernel_launch(void* const* d_in, const int* in_sizes, int n_in,
                              void* d_out, int out_size, void* d_ws, size_t ws_size,
                              hipStream_t stream) {
  (void)in_sizes; (void)n_in; (void)out_size; (void)ws_size;
  const float* x     = (const float*)d_in[0];
  const float* gamma = (const float*)d_in[1];
  float*       out   = (float*)d_out;
  _Float16*    w16   = (_Float16*)d_ws;

  // allow >64KB dynamic LDS (host API, graph-capture safe: not a stream op)
  static_assert(LDS_BYTES <= 320 * 1024, "LDS budget");
  (void)hipFuncSetAttribute((const void*)axial_pass,
                            hipFuncAttributeMaxDynamicSharedMemorySize, LDS_BYTES);

  const int QKV_N = 3 * CCH * CCH;   // 196608
  const int O_N   = CCH * CCH;       // 65536
  const int PER   = QKV_N + O_N;     // 262144 halfs/pass -> 1.5MB total in d_ws

  // repack fp32 weights -> fp16 (per pass: wqkv then wo)
  for (int p = 0; p < 3; ++p) {
    const float* wq = (const float*)d_in[2 + 6 * p + 2];
    const float* wo = (const float*)d_in[2 + 6 * p + 4];
    hipLaunchKernelGGL(repack_f16, dim3((QKV_N + 255) / 256), dim3(256), 0, stream,
                       wq, w16 + (size_t)p * PER, QKV_N);
    hipLaunchKernelGGL(repack_f16, dim3((O_N + 255) / 256), dim3(256), 0, stream,
                       wo, w16 + (size_t)p * PER + QKV_N, O_N);
  }

  // axis passes: depth (x->out), then height and width in-place on out.
  // token stride / fixed-coord strides in elements of the (b,c,d,h,w) layout
  const long st_t[3] = {2304, 48, 1};      // d, h, w
  const long st_a[3] = {48, 2304, 2304};   // fixed coord A: h, d, d
  const long st_b[3] = {1, 1, 48};         // fixed coord B: w, w, h
  for (int p = 0; p < 3; ++p) {
    const float* src  = (p == 0) ? x : out;
    const float* lnw  = (const float*)d_in[2 + 6 * p + 0];
    const float* lnb  = (const float*)d_in[2 + 6 * p + 1];
    const float* bqkv = (const float*)d_in[2 + 6 * p + 3];
    const float* bo   = (const float*)d_in[2 + 6 * p + 5];
    hipLaunchKernelGGL(axial_pass, dim3(2 * 48 * 48), dim3(256), LDS_BYTES, stream,
                       src, out, gamma, lnw, lnb,
                       w16 + (size_t)p * PER, bqkv,
                       w16 + (size_t)p * PER + QKV_N, bo,
                       st_t[p], st_a[p], st_b[p]);
  }
}
